// CrossAttention_9534827397696
// MI455X (gfx1250) — compile-verified, matrix operand order checked by reading, not run on hardware
//
#include <hip/hip_runtime.h>
#include <hip/hip_bf16.h>

// MI455X (gfx1250) fused linear cross-attention.
// Heavy op: Q = to_q_w (512x512) x x[b] (512x4096) in bf16 WMMA, f32 accum.
// Everything else collapses to rank-1 algebra (see analysis).

typedef __attribute__((ext_vector_type(16))) __bf16 v16bf;
typedef __attribute__((ext_vector_type(8)))  __bf16 v8bf;
typedef __attribute__((ext_vector_type(8)))  float  v8f;

#define HEADS  8
#define DHEAD  64
#define CCH    512          // channels c
#define HID    512          // heads*dim_head
#define DIMY   768
#define NPIX   4096         // 64*64
#define NB     16
#define TILE_N 64
#define KSTR   520          // bf16 K-stride in LDS (mult of 8 -> 16B aligned frags)
#define QSTR   68
#define GN_EPS 1e-5f

// workspace layout (float offsets)
#define WS_KY    0            // 16*512
#define WS_VY    8192         // 16*512
#define WS_SK    16384        // 512
#define WS_SV    16896        // 512
#define WS_CTX   17408        // 16*512
#define WS_W2    25600        // 512*8
#define WS_PART  29696        // 16*64*2
#define WS_STAT  31744        // 16*2

// ---------------- small setup kernels ----------------

__global__ __launch_bounds__(256) void k_kyvy(const float* __restrict__ y,
                                              const float* __restrict__ k_w,
                                              const float* __restrict__ v_w,
                                              float* __restrict__ ws) {
    __shared__ float ylds[DIMY];
    int b = blockIdx.x, t = threadIdx.x;
    for (int i = t; i < DIMY; i += 256) ylds[i] = y[b * DIMY + i];
    __syncthreads();
    for (int o = t; o < CCH; o += 256) {
        float sk = 0.f, sv = 0.f;
        const float* kr = k_w + (size_t)o * DIMY;
        const float* vr = v_w + (size_t)o * DIMY;
        for (int d = 0; d < DIMY; ++d) { float yv = ylds[d]; sk += yv * kr[d]; sv += yv * vr[d]; }
        ws[WS_KY + b * CCH + o] = sk;
        ws[WS_VY + b * CCH + o] = sv;
    }
}

__global__ __launch_bounds__(256) void k_sksv(const float* __restrict__ to_k_w,
                                              const float* __restrict__ to_v_w,
                                              float* __restrict__ ws) {
    int j = blockIdx.x * 256 + threadIdx.x;
    if (j < HID) {
        float a = 0.f, bsum = 0.f;
        const float* kr = to_k_w + (size_t)j * CCH;
        const float* vr = to_v_w + (size_t)j * CCH;
        for (int c = 0; c < CCH; ++c) { a += kr[c]; bsum += vr[c]; }
        ws[WS_SK + j] = a;
        ws[WS_SV + j] = bsum;
    }
}

__global__ __launch_bounds__(256) void k_ctx(float* __restrict__ ws) {
    __shared__ float kyl[CCH], vyl[CCH];
    int b = blockIdx.x, t = threadIdx.x;
    for (int i = t; i < CCH; i += 256) { kyl[i] = ws[WS_KY + b * CCH + i]; vyl[i] = ws[WS_VY + b * CCH + i]; }
    __syncthreads();
    for (int j = t; j < HID; j += 256) {
        float s = ws[WS_SK + j];
        float m = -1e30f;
        for (int n = 0; n < CCH; ++n) m = fmaxf(m, s * kyl[n]);
        float Z = 0.f, S = 0.f;
        for (int n = 0; n < CCH; ++n) {
            float e = __expf(s * kyl[n] - m);
            Z += e; S += e * vyl[n];
        }
        ws[WS_CTX + b * HID + j] = S / Z;
    }
}

__global__ __launch_bounds__(256) void k_w2(const float* __restrict__ out_w,
                                            float* __restrict__ ws) {
    int idx = blockIdx.x * 256 + threadIdx.x;   // grid 16 -> 4096
    if (idx < HID * HEADS) {
        int o = idx >> 3, h = idx & 7;
        float s = 0.f;
        const float* wr  = out_w + (size_t)o * HID + h * DHEAD;
        const float* svp = ws + WS_SV + h * DHEAD;
        for (int e = 0; e < DHEAD; ++e) s += wr[e] * svp[e];
        ws[WS_W2 + o * HEADS + h] = s;
    }
}

// ---------------- main fused kernel ----------------
// grid: (64 pixel-tiles, 16 batches), 256 threads (8 wave32)

__device__ inline v16bf ld_frag(const __bf16* p) {
    // elements k0..k0+7 then k0+16..k0+23 (two 16B LDS reads)
    const v8bf* q = (const v8bf*)p;
    v8bf lo = q[0];
    v8bf hi = q[2];
    return __builtin_shufflevector(lo, hi, 0,1,2,3,4,5,6,7,8,9,10,11,12,13,14,15);
}

__global__ __launch_bounds__(256) void k_main(const float* __restrict__ x,
                                              const float* __restrict__ to_q_w,
                                              const float* __restrict__ out_b,
                                              const float* __restrict__ ws,
                                              float* __restrict__ out,
                                              float* __restrict__ part) {
    extern __shared__ char smem[];
    __bf16* ldsA  = (__bf16*)smem;                               // 64*520*2   = 66560 B
    __bf16* ldsB  = (__bf16*)(smem + 66560);                     // 64*520*2   = 66560 B
    float*  Qbuf  = (float*)(smem + 133120);                     // 64*68*4    = 17408 B
    float*  sTile = (float*)(smem + 150528);                     // 8*64*4     = 2048 B
    float*  ctxl  = (float*)(smem + 152576);                     // 64*4       = 256 B
    float*  red   = (float*)(smem + 152832);                     // 256*4      = 1024 B
    float*  W2l   = Qbuf;                                        // reuse after head loop

    const int b    = blockIdx.y;
    const int pix0 = blockIdx.x * TILE_N;
    const int tid  = threadIdx.x;
    const int lane = tid & 31;
    const int wv   = tid >> 5;

    // stage x tile -> ldsB transposed (ldsB[n*KSTR + c]) as bf16
    const float* xb = x + (size_t)b * CCH * NPIX;
    for (int idx = tid; idx < CCH * TILE_N; idx += 256) {
        int c = idx >> 6, n = idx & 63;
        ldsB[n * KSTR + c] = (__bf16)xb[(size_t)c * NPIX + pix0 + n];
    }

    // wave tile assignment: 4 row-groups x 2 column-pairs; A frag reused twice
    const int m0    = (wv >> 1) * 16;
    const int n0    = (wv & 1) * 32;
    const int rowA  = m0 + (lane & 15);
    const int colB0 = n0 + (lane & 15);
    const int colB1 = colB0 + 16;
    const int ksel  = (lane >> 4) * 8;
    const int crow  = m0 + ((lane >> 4) << 3);
    const int ccol  = n0 + (lane & 15);

    for (int h = 0; h < HEADS; ++h) {
        __syncthreads();   // prev head's ldsA reads and Qbuf consumers done
        const float* aw = to_q_w + (size_t)h * DHEAD * CCH;
        for (int idx = tid; idx < DHEAD * CCH; idx += 256) {
            int r = idx >> 9, c = idx & 511;
            ldsA[r * KSTR + c] = (__bf16)aw[r * CCH + c];
        }
        if (tid < DHEAD) ctxl[tid] = ws[WS_CTX + b * HID + h * DHEAD + tid];
        __syncthreads();

        v8f acc0 = {}; v8f acc1 = {};
        for (int kb = 0; kb < CCH; kb += 32) {
            int k0 = kb + ksel;
            v16bf a  = ld_frag(ldsA + rowA  * KSTR + k0);
            v16bf b0 = ld_frag(ldsB + colB0 * KSTR + k0);
            v16bf b1 = ld_frag(ldsB + colB1 * KSTR + k0);
            acc0 = __builtin_amdgcn_wmma_f32_16x16x32_bf16(false, a, false, b0, (short)0, acc0, false, false);
            acc1 = __builtin_amdgcn_wmma_f32_16x16x32_bf16(false, a, false, b1, (short)0, acc1, false, false);
        }
        #pragma unroll
        for (int g = 0; g < 8; ++g) {
            Qbuf[(crow + g) * QSTR + ccol]      = acc0[g];
            Qbuf[(crow + g) * QSTR + ccol + 16] = acc1[g];
        }
        __syncthreads();

        // fused softmax over d + dot with ctx_v, per column
        if (tid < TILE_N) {
            const int n = tid;
            float m = -1e30f;
            for (int d = 0; d < DHEAD; ++d) m = fmaxf(m, Qbuf[d * QSTR + n]);
            float Z = 0.f, S = 0.f;
            for (int d = 0; d < DHEAD; ++d) {
                float e = __expf(Qbuf[d * QSTR + n] - m);
                Z += e; S += ctxl[d] * e;
            }
            sTile[h * TILE_N + n] = S / Z * 0.125f;   // scale = 1/sqrt(64)
        }
    }
    __syncthreads();

    // output mix: final[o,n] = out_b[o] + sum_h W2[o,h]*s[h,n]
    for (int i = tid; i < HID * HEADS; i += 256) W2l[i] = ws[WS_W2 + i];
    __syncthreads();

    float lsum = 0.f, lsq = 0.f;
    float* ob = out + (size_t)b * CCH * NPIX + pix0;
    for (int it = 0; it < (CCH * TILE_N) / 256; ++it) {
        int idx = tid + it * 256;
        int o = idx >> 6, n = idx & 63;
        float v = out_b[o];
        #pragma unroll
        for (int h = 0; h < HEADS; ++h) v += W2l[o * HEADS + h] * sTile[h * TILE_N + n];
        ob[(size_t)o * NPIX + n] = v;
        lsum += v; lsq += v * v;
    }

    // deterministic per-block partial sums for GroupNorm
    red[tid] = lsum; __syncthreads();
    for (int s = 128; s > 0; s >>= 1) { if (tid < s) red[tid] += red[tid + s]; __syncthreads(); }
    float bsum = red[0]; __syncthreads();
    red[tid] = lsq; __syncthreads();
    for (int s = 128; s > 0; s >>= 1) { if (tid < s) red[tid] += red[tid + s]; __syncthreads(); }
    if (tid == 0) {
        part[(b * 64 + blockIdx.x) * 2 + 0] = bsum;
        part[(b * 64 + blockIdx.x) * 2 + 1] = red[0];
    }
}

__global__ __launch_bounds__(64) void k_stat(const float* __restrict__ part,
                                             float* __restrict__ stat) {
    __shared__ float r0[64], r1[64];
    int b = blockIdx.x, t = threadIdx.x;
    r0[t] = part[(b * 64 + t) * 2 + 0];
    r1[t] = part[(b * 64 + t) * 2 + 1];
    __syncthreads();
    for (int s = 32; s > 0; s >>= 1) {
        if (t < s) { r0[t] += r0[t + s]; r1[t] += r1[t + s]; }
        __syncthreads();
    }
    if (t == 0) { stat[b * 2 + 0] = r0[0]; stat[b * 2 + 1] = r1[0]; }
}

__global__ __launch_bounds__(256) void k_gn(float* __restrict__ out,
                                            const float* __restrict__ stat,
                                            const float* __restrict__ g,
                                            const float* __restrict__ bb) {
    size_t idx = (size_t)blockIdx.x * 256 + threadIdx.x;
    int b = (int)(idx >> 21);          // 512*4096 = 2^21 elems per batch
    int o = (int)((idx >> 12) & 511);
    const float NE = (float)CCH * (float)NPIX;
    float mu  = stat[b * 2 + 0] / NE;
    float var = stat[b * 2 + 1] / NE - mu * mu;
    float inv = rsqrtf(var + GN_EPS);
    float v = out[idx];
    out[idx] = (v - mu) * inv * g[o] + bb[o];
}

// ---------------- launcher ----------------

extern "C" void kernel_launch(void* const* d_in, const int* in_sizes, int n_in,
                              void* d_out, int out_size, void* d_ws, size_t ws_size,
                              hipStream_t stream) {
    const float* x      = (const float*)d_in[0];
    const float* y      = (const float*)d_in[1];
    const float* k_w    = (const float*)d_in[2];
    const float* v_w    = (const float*)d_in[3];
    const float* to_q_w = (const float*)d_in[4];
    const float* to_k_w = (const float*)d_in[5];
    const float* to_v_w = (const float*)d_in[6];
    const float* out_w  = (const float*)d_in[7];
    const float* out_b  = (const float*)d_in[8];
    const float* gn_g   = (const float*)d_in[9];
    const float* gn_b   = (const float*)d_in[10];
    float* out = (float*)d_out;
    float* ws  = (float*)d_ws;

    k_kyvy<<<NB, 256, 0, stream>>>(y, k_w, v_w, ws);
    k_sksv<<<2, 256, 0, stream>>>(to_k_w, to_v_w, ws);
    k_ctx <<<NB, 256, 0, stream>>>(ws);
    k_w2  <<<16, 256, 0, stream>>>(out_w, ws);

    dim3 g5(NPIX / TILE_N, NB);
    size_t smem = 66560u * 2 + 17408u + 2048u + 256u + 1024u;  // 153856 B of 320KB WGP LDS
    k_main<<<g5, 256, smem, stream>>>(x, to_q_w, out_b, ws, out, ws + WS_PART);

    k_stat<<<NB, 64, 0, stream>>>(ws + WS_PART, ws + WS_STAT);

    int total = out_size;                       // 33554432
    k_gn<<<total / 256, 256, 0, stream>>>(out, ws + WS_STAT, gn_g, gn_b);
}
